// MultiHeadSelfAttention_12678743458070
// MI455X (gfx1250) — matrix operand order repaired and checked by reading, not compiled, not run on hardware
//
#include <hip/hip_runtime.h>
#include <hip/hip_bf16.h>

typedef _Float16 v16h __attribute__((ext_vector_type(16)));
typedef _Float16 v8h  __attribute__((ext_vector_type(8)));
typedef float    v8f  __attribute__((ext_vector_type(8)));

#define D_MODEL  1024
#define N_HEADS  16
#define HEAD_DIM 64
#define SEQ      2048
#define BATCH    4
#define M_TOTAL  (BATCH * SEQ)   // 8192

// ---------------------------------------------------------------- helpers
__device__ __forceinline__ v16h pack16(v8h lo, v8h hi) {
    v16h r;
#pragma unroll
    for (int i = 0; i < 8; ++i) { r[i] = lo[i]; r[i + 8] = hi[i]; }
    return r;
}

__device__ __forceinline__ v8f wmma16(v16h a, v16h b, v8f c) {
    return __builtin_amdgcn_wmma_f32_16x16x32_f16(
        false, a, false, b, (short)0, c, false, false);
}

// ---------------------------------------------------------------- f32 -> f16 (8 elems/thread)
__global__ void cvt_f32_f16(const float* __restrict__ in,
                            _Float16* __restrict__ out, int n8) {
    int i = blockIdx.x * blockDim.x + threadIdx.x;
    if (i < n8) {
        const float4* p = (const float4*)in + (size_t)i * 2;
        float4 a = p[0], b = p[1];
        v8h o;
        o[0] = (_Float16)a.x; o[1] = (_Float16)a.y;
        o[2] = (_Float16)a.z; o[3] = (_Float16)a.w;
        o[4] = (_Float16)b.x; o[5] = (_Float16)b.y;
        o[6] = (_Float16)b.z; o[7] = (_Float16)b.w;
        *(v8h*)(out + (size_t)i * 8) = o;
    }
}

// ---------------------------------------------------------------- GEMM
// y[m,n] = sum_k A[m,k] * W[n,k] + bias[n]
// Block tile 256(M) x 64(N), K-step 32, double-buffered LDS.
// 256 threads = 8 waves; wave owns 32x64 (2 M-frags x 4 N-frags = 8 WMMA/K-step).
// mode 0: f16 [B,H,S,hd] (Q,K) | mode 1: f16 [B,H,hd,S] (V) | mode 2: f32 [M,D]
#define GM   256
#define GN   64
#define GK   32
#define ASTR 40   // 32 + 8 f16 pad: 16B aligned rows, conflict-free fragment reads
#define BSTR 40

__global__ __launch_bounds__(256) void gemm_f16_wmma(
    const _Float16* __restrict__ A, const _Float16* __restrict__ W,
    const float* __restrict__ bias, void* __restrict__ out, int mode)
{
    __shared__ _Float16 As[2][GM * ASTR];
    __shared__ _Float16 Bs[2][GN * BSTR];

    const int tid  = threadIdx.x;
    const int wave = tid >> 5;
    const int lane = tid & 31;
    const int half = lane >> 4;
    const int ln   = lane & 15;
    const int mbase = blockIdx.x * GM;
    const int nbase = blockIdx.y * GN;

    // per-thread staging coordinates (A: 4 chunks, B: 1 chunk of 8 f16)
    const int br = tid >> 2, bp = tid & 3;

    v8f acc[2][4];
#pragma unroll
    for (int mt = 0; mt < 2; ++mt)
#pragma unroll
        for (int nt = 0; nt < 4; ++nt) acc[mt][nt] = (v8f){};

    // ---- stage tile k0=0 into buffer 0
    {
#pragma unroll
        for (int c = 0; c < 4; ++c) {
            int chunk = tid + c * 256;
            int r = chunk >> 2, p = chunk & 3;
            *(v8h*)(As[0] + r * ASTR + p * 8) =
                *(const v8h*)(A + (size_t)(mbase + r) * D_MODEL + p * 8);
        }
        *(v8h*)(Bs[0] + br * BSTR + bp * 8) =
            *(const v8h*)(W + (size_t)(nbase + br) * D_MODEL + bp * 8);
    }
    __syncthreads();

    int cur = 0;
    for (int k0 = 0; k0 < D_MODEL; k0 += GK) {
        const int  nxt     = cur ^ 1;
        const bool hasNext = (k0 + GK) < D_MODEL;

        // ---- issue next-tile global loads early (latency hides under WMMAs)
        v8h ar[4], brv;
        if (hasNext) {
#pragma unroll
            for (int c = 0; c < 4; ++c) {
                int chunk = tid + c * 256;
                int r = chunk >> 2, p = chunk & 3;
                ar[c] = *(const v8h*)(A + (size_t)(mbase + r) * D_MODEL + k0 + GK + p * 8);
            }
            brv = *(const v8h*)(W + (size_t)(nbase + br) * D_MODEL + k0 + GK + bp * 8);
        }

        // ---- fragment loads: everything live, single wait, then 8 WMMAs
        const _Float16* Ac = As[cur];
        const _Float16* Bc = Bs[cur];
        v16h afr[2];
#pragma unroll
        for (int mt = 0; mt < 2; ++mt) {
            const int arow = wave * 32 + mt * 16 + ln;
            v8h lo = *(const v8h*)(Ac + arow * ASTR + half * 8);
            v8h hi = *(const v8h*)(Ac + arow * ASTR + 16 + half * 8);
            afr[mt] = pack16(lo, hi);
        }
        v16h bfr[4];
#pragma unroll
        for (int nt = 0; nt < 4; ++nt) {
            const int brow = nt * 16 + ln;
            v8h b0 = *(const v8h*)(Bc + brow * BSTR + half * 8);
            v8h b1 = *(const v8h*)(Bc + brow * BSTR + 16 + half * 8);
            bfr[nt] = pack16(b0, b1);
        }
#pragma unroll
        for (int mt = 0; mt < 2; ++mt)
#pragma unroll
            for (int nt = 0; nt < 4; ++nt)
                acc[mt][nt] = wmma16(afr[mt], bfr[nt], acc[mt][nt]);

        // ---- commit next tile to the other buffer
        if (hasNext) {
#pragma unroll
            for (int c = 0; c < 4; ++c) {
                int chunk = tid + c * 256;
                int r = chunk >> 2, p = chunk & 3;
                *(v8h*)(As[nxt] + r * ASTR + p * 8) = ar[c];
            }
            *(v8h*)(Bs[nxt] + br * BSTR + bp * 8) = brv;
        }
        __syncthreads();
        cur = nxt;
    }

    // ---- epilogue: C layout row = half*8 + e, col = ln
#pragma unroll
    for (int mt = 0; mt < 2; ++mt)
#pragma unroll
        for (int nt = 0; nt < 4; ++nt) {
            const int n  = nbase + nt * 16 + ln;
            const float bv = bias[n];
#pragma unroll
            for (int e = 0; e < 8; ++e) {
                const int m = mbase + wave * 32 + mt * 16 + half * 8 + e;
                const float v = acc[mt][nt][e] + bv;
                if (mode == 2) {
                    ((float*)out)[(size_t)m * D_MODEL + n] = v;
                } else {
                    const int b = m >> 11, s = m & (SEQ - 1);
                    const int h = n >> 6, d = n & (HEAD_DIM - 1);
                    const size_t idx = (mode == 0)
                        ? ((size_t)((b * N_HEADS + h) * SEQ + s)) * HEAD_DIM + d
                        : ((size_t)((b * N_HEADS + h) * HEAD_DIM + d)) * SEQ + s;
                    ((_Float16*)out)[idx] = (_Float16)v;
                }
            }
        }
}

// ---------------------------------------------------------------- flash attention
// 4 waves / 64 query rows per workgroup; 64-key blocks, double-buffered K/V.
#define QB   64
#define KB   64
#define NKB  (SEQ / KB)
#define QSTR 72   // 64 + 8 f16 pad

__global__ __launch_bounds__(128) void flash_attn_wmma(
    const _Float16* __restrict__ Q,   // [B,H,S,hd]
    const _Float16* __restrict__ K,   // [B,H,S,hd]
    const _Float16* __restrict__ Vt,  // [B,H,hd,S]
    _Float16* __restrict__ O)         // [B*S, D_MODEL] merged heads, f16
{
    __shared__ _Float16 Qs[QB * QSTR];
    __shared__ _Float16 Ks[2][KB * QSTR];
    __shared__ _Float16 Vs[2][HEAD_DIM * QSTR];   // [hd][key]
    __shared__ _Float16 Ps[4 * 16 * KB];

    const int tid  = threadIdx.x;
    const int wave = tid >> 5;
    const int lane = tid & 31;
    const int half = lane >> 4;
    const int ln   = lane & 15;

    const int qb = blockIdx.x;
    const int h  = blockIdx.y;
    const int b  = blockIdx.z;

    const size_t headoff = (size_t)(b * N_HEADS + h) * SEQ * HEAD_DIM;
    const _Float16* Qg = Q  + headoff + (size_t)qb * QB * HEAD_DIM;
    const _Float16* Kg = K  + headoff;
    const _Float16* Vg = Vt + headoff;            // [hd][SEQ]

    // ---- stage Q tile + key-block 0 into buffer 0
#pragma unroll
    for (int c = 0; c < 4; ++c) {
        int chunk = tid + c * 128;
        int r = chunk >> 3, p = chunk & 7;
        *(v8h*)(Qs + r * QSTR + p * 8) = *(const v8h*)(Qg + r * HEAD_DIM + p * 8);
        *(v8h*)(Ks[0] + r * QSTR + p * 8) =
            *(const v8h*)(Kg + (size_t)r * HEAD_DIM + p * 8);
        *(v8h*)(Vs[0] + r * QSTR + p * 8) =
            *(const v8h*)(Vg + (size_t)r * SEQ + p * 8);
    }
    __syncthreads();

    v8f oacc[4];
#pragma unroll
    for (int i = 0; i < 4; ++i) oacc[i] = (v8f){};
    float mrow[8], lrow[8];
#pragma unroll
    for (int e = 0; e < 8; ++e) { mrow[e] = -1e30f; lrow[e] = 0.0f; }

    _Float16* Pw = Ps + wave * 16 * KB;

    int cur = 0;
    for (int kb = 0; kb < NKB; ++kb) {
        const int  nxt     = cur ^ 1;
        const bool hasNext = (kb + 1) < NKB;

        // ---- issue next K/V block global loads (hide under score WMMAs)
        v8h kr[4], vr[4];
        if (hasNext) {
#pragma unroll
            for (int c = 0; c < 4; ++c) {
                int chunk = tid + c * 128;
                int r = chunk >> 3, p = chunk & 7;
                kr[c] = *(const v8h*)(Kg + (size_t)((kb + 1) * KB + r) * HEAD_DIM + p * 8);
                vr[c] = *(const v8h*)(Vg + (size_t)r * SEQ + (kb + 1) * KB + p * 8);
            }
        }

        // ---- scores: S[16 x 64] = Q(wave rows) x K_block^T, batched fragments
        const _Float16* Kc = Ks[cur];
        v8f sfr[4];
#pragma unroll
        for (int j = 0; j < 4; ++j) sfr[j] = (v8f){};
#pragma unroll
        for (int kk = 0; kk < 2; ++kk) {
            const int arow = wave * 16 + ln;
            v8h alo = *(const v8h*)(Qs + arow * QSTR + kk * 32 + half * 8);
            v8h ahi = *(const v8h*)(Qs + arow * QSTR + kk * 32 + 16 + half * 8);
            v16h af = pack16(alo, ahi);
            v16h bfr[4];
#pragma unroll
            for (int j = 0; j < 4; ++j) {
                const int brow = j * 16 + ln;
                v8h b0 = *(const v8h*)(Kc + brow * QSTR + kk * 32 + half * 8);
                v8h b1 = *(const v8h*)(Kc + brow * QSTR + kk * 32 + 16 + half * 8);
                bfr[j] = pack16(b0, b1);
            }
#pragma unroll
            for (int j = 0; j < 4; ++j) sfr[j] = wmma16(af, bfr[j], sfr[j]);
        }

        // ---- online softmax (row = 16 lanes of one half)
        float alpha[8];
#pragma unroll
        for (int e = 0; e < 8; ++e) {
            float t = fmaxf(fmaxf(sfr[0][e], sfr[1][e]),
                            fmaxf(sfr[2][e], sfr[3][e])) * 0.125f;
#pragma unroll
            for (int off = 1; off < 16; off <<= 1)
                t = fmaxf(t, __shfl_xor(t, off, 16));
            const float mnew = fmaxf(mrow[e], t);
            alpha[e] = __expf(mrow[e] - mnew);
            mrow[e]  = mnew;
        }
        float rs[8];
#pragma unroll
        for (int e = 0; e < 8; ++e) rs[e] = 0.0f;
#pragma unroll
        for (int j = 0; j < 4; ++j)
#pragma unroll
            for (int e = 0; e < 8; ++e) {
                float p = __expf(sfr[j][e] * 0.125f - mrow[e]);
                sfr[j][e] = p;
                rs[e] += p;
            }
#pragma unroll
        for (int e = 0; e < 8; ++e) {
            float t = rs[e];
#pragma unroll
            for (int off = 1; off < 16; off <<= 1)
                t += __shfl_xor(t, off, 16);
            lrow[e] = lrow[e] * alpha[e] + t;
        }
#pragma unroll
        for (int nt = 0; nt < 4; ++nt)
#pragma unroll
            for (int e = 0; e < 8; ++e) oacc[nt][e] *= alpha[e];

        // ---- reshape P (C layout -> row-major LDS)
#pragma unroll
        for (int j = 0; j < 4; ++j)
#pragma unroll
            for (int e = 0; e < 8; ++e)
                Pw[(half * 8 + e) * KB + j * 16 + ln] = (_Float16)sfr[j][e];

        // ---- commit next K/V block to the other buffer
        if (hasNext) {
#pragma unroll
            for (int c = 0; c < 4; ++c) {
                int chunk = tid + c * 128;
                int r = chunk >> 3, p = chunk & 7;
                *(v8h*)(Ks[nxt] + r * QSTR + p * 8) = kr[c];
                *(v8h*)(Vs[nxt] + r * QSTR + p * 8) = vr[c];
            }
        }
        __syncthreads();   // Ps visible; next K/V staged

        // ---- O += P[16 x 64] x V[64 x 64]
        const _Float16* Vc = Vs[cur];
#pragma unroll
        for (int kk = 0; kk < 2; ++kk) {
            v8h alo = *(const v8h*)(Pw + ln * KB + kk * 32 + half * 8);
            v8h ahi = *(const v8h*)(Pw + ln * KB + kk * 32 + 16 + half * 8);
            v16h af = pack16(alo, ahi);
            v16h bfr[4];
#pragma unroll
            for (int nt = 0; nt < 4; ++nt) {
                const int brow = nt * 16 + ln;
                v8h b0 = *(const v8h*)(Vc + brow * QSTR + kk * 32 + half * 8);
                v8h b1 = *(const v8h*)(Vc + brow * QSTR + kk * 32 + 16 + half * 8);
                bfr[nt] = pack16(b0, b1);
            }
#pragma unroll
            for (int nt = 0; nt < 4; ++nt)
                oacc[nt] = wmma16(af, bfr[nt], oacc[nt]);
        }
        __syncthreads();   // PV done before next iteration overwrites Vs[cur]
        cur = nxt;
    }

    // ---- normalize + store merged heads f16 [B*S, D_MODEL]
#pragma unroll
    for (int e = 0; e < 8; ++e) {
        const float inv = 1.0f / lrow[e];
        const int m = b * SEQ + qb * QB + wave * 16 + half * 8 + e;
#pragma unroll
        for (int nt = 0; nt < 4; ++nt) {
            const int d = h * HEAD_DIM + nt * 16 + ln;
            O[(size_t)m * D_MODEL + d] = (_Float16)(oacc[nt][e] * inv);
        }
    }
}

// ---------------------------------------------------------------- launch
extern "C" void kernel_launch(void* const* d_in, const int* in_sizes, int n_in,
                              void* d_out, int out_size, void* d_ws, size_t ws_size,
                              hipStream_t stream) {
    const float* x  = (const float*)d_in[0];
    const float* Wq = (const float*)d_in[1];
    const float* bq = (const float*)d_in[2];
    const float* Wk = (const float*)d_in[3];
    const float* bk = (const float*)d_in[4];
    const float* Wv = (const float*)d_in[5];
    const float* bv = (const float*)d_in[6];
    const float* Wo = (const float*)d_in[7];
    const float* bo = (const float*)d_in[8];

    char* ws = (char*)d_ws;
    const size_t XB = (size_t)M_TOTAL * D_MODEL * sizeof(_Float16);  // 16 MB
    const size_t WB = (size_t)D_MODEL * D_MODEL * sizeof(_Float16);  //  2 MB
    _Float16* xh  = (_Float16*)ws;            ws += XB;
    _Float16* wqh = (_Float16*)ws;            ws += WB;
    _Float16* wkh = (_Float16*)ws;            ws += WB;
    _Float16* wvh = (_Float16*)ws;            ws += WB;
    _Float16* woh = (_Float16*)ws;            ws += WB;
    _Float16* qh  = (_Float16*)ws;            ws += XB;   // [B,H,S,hd]
    _Float16* kh  = (_Float16*)ws;            ws += XB;   // [B,H,S,hd]
    _Float16* vth = (_Float16*)ws;            ws += XB;   // [B,H,hd,S]
    _Float16* ah  = (_Float16*)ws;            ws += XB;   // [B*S, D]

    const int NX8 = M_TOTAL * D_MODEL / 8;
    const int NW8 = D_MODEL * D_MODEL / 8;
    cvt_f32_f16<<<(NX8 + 255) / 256, 256, 0, stream>>>(x,  xh,  NX8);
    cvt_f32_f16<<<(NW8 + 255) / 256, 256, 0, stream>>>(Wq, wqh, NW8);
    cvt_f32_f16<<<(NW8 + 255) / 256, 256, 0, stream>>>(Wk, wkh, NW8);
    cvt_f32_f16<<<(NW8 + 255) / 256, 256, 0, stream>>>(Wv, wvh, NW8);
    cvt_f32_f16<<<(NW8 + 255) / 256, 256, 0, stream>>>(Wo, woh, NW8);

    dim3 gg(M_TOTAL / GM, D_MODEL / GN);      // 32 x 16
    gemm_f16_wmma<<<gg, 256, 0, stream>>>(xh, wqh, bq, qh,  0);
    gemm_f16_wmma<<<gg, 256, 0, stream>>>(xh, wkh, bk, kh,  0);
    gemm_f16_wmma<<<gg, 256, 0, stream>>>(xh, wvh, bv, vth, 1);

    dim3 ga(SEQ / QB, N_HEADS, BATCH);        // 32 x 16 x 4
    flash_attn_wmma<<<ga, 128, 0, stream>>>(qh, kh, vth, ah);

    gemm_f16_wmma<<<gg, 256, 0, stream>>>(ah, woh, bo, d_out, 2);
}
